// ConceptRoutingHead_4131758538929
// MI455X (gfx1250) — matrix operand order repaired and checked by reading, not compiled, not run on hardware
//
#include <hip/hip_runtime.h>

// ---------------- types ----------------
typedef __bf16 v16bf __attribute__((ext_vector_type(16)));
typedef float  v8f   __attribute__((ext_vector_type(8)));
typedef float  f32x4 __attribute__((ext_vector_type(4)));
typedef unsigned int u32x4 __attribute__((ext_vector_type(4)));
typedef unsigned int u32x2 __attribute__((ext_vector_type(2)));

union FragBF { v16bf v; u32x4 q[2]; };

static constexpr int Bn = 2048;   // batch
static constexpr int Cn = 512;    // concepts
static constexpr int Dn = 256;    // proj dim
static constexpr int PD = 48;     // padded LDS K-stride (bf16 elems; 96B row: 16B-aligned, bank-skewed)

// ---- CDNA5 async global->LDS (ASYNCcnt path, no VGPR round-trip) ----
__device__ __forceinline__ void async_b128(unsigned lds_off, const void* gptr) {
  asm volatile("global_load_async_to_lds_b128 %0, %1, off"
               :: "v"(lds_off), "v"(gptr) : "memory");
}
__device__ __forceinline__ void async_b64(unsigned lds_off, const void* gptr) {
  asm volatile("global_load_async_to_lds_b64 %0, %1, off"
               :: "v"(lds_off), "v"(gptr) : "memory");
}
__device__ __forceinline__ void wait_async0() {
  asm volatile("s_wait_asynccnt 0" ::: "memory");
}
__device__ __forceinline__ unsigned lds_addr(const void* p) {
  return (unsigned)(unsigned long long)p;  // LDS aperture: addr[31:0] = LDS offset
}

// ---------------- fp32 -> bf16 convert (4 elems/thread) ----------------
__global__ __launch_bounds__(256) void cvt_f32_bf16(const float* __restrict__ s,
                                                    __bf16* __restrict__ d) {
  int i = (blockIdx.x * 256 + threadIdx.x) * 4;
  f32x4 v = *(const f32x4*)(s + i);
  union { __bf16 h[4]; u32x2 u; } p;
  p.h[0] = (__bf16)v[0]; p.h[1] = (__bf16)v[1];
  p.h[2] = (__bf16)v[2]; p.h[3] = (__bf16)v[3];
  *(u32x2*)(d + i) = p.u;
}

// ---------------- L2-normalize rows of U (fp32 [rows,256] -> bf16) ----------------
__global__ __launch_bounds__(256) void l2rows_kernel(const float* __restrict__ src,
                                                     __bf16* __restrict__ dst) {
  int lane = threadIdx.x & 31, wave = threadIdx.x >> 5;
  int row = blockIdx.x * 8 + wave;
  const float* p = src + row * Dn;
  float v[8], ss = 0.f;
#pragma unroll
  for (int i = 0; i < 8; ++i) { v[i] = p[lane + i * 32]; ss += v[i] * v[i]; }
#pragma unroll
  for (int off = 16; off > 0; off >>= 1) ss += __shfl_xor(ss, off, 32);
  float inv = 1.f / (sqrtf(ss) + 1e-6f);
#pragma unroll
  for (int i = 0; i < 8; ++i) dst[row * Dn + lane + i * 32] = (__bf16)(v[i] * inv);
}

// ---------------- projection GEMM (Y = f @ W^T + b) + LayerNorm + L2 ----------------
// block = 256 thr (8 waves), tile = 32 rows x 256 cols (full D). Wave: 16x64 (4 wmma frags).
// Double-buffered LDS, filled by async-to-LDS while WMMAs consume the other buffer.
__global__ __launch_bounds__(256) void proj_ln_kernel(
    const __bf16* __restrict__ A, int K, const __bf16* __restrict__ W,
    const float* __restrict__ bias, const float* __restrict__ gamma,
    const float* __restrict__ beta, __bf16* __restrict__ Zbf,
    __bf16* __restrict__ Znb, int mod) {
  static constexpr int ABYT = 32 * PD * 2;     // 3072
  static constexpr int BUF  = ABYT + 256 * PD * 2;  // 27648 per stage buffer
  __shared__ __align__(16) unsigned char smem[2 * BUF];  // 55296; Ys aliases base
  float* Ys = (float*)smem;  // [32][264] = 33792 bytes (used after GEMM)

  const int tid = threadIdx.x, lane = tid & 31, wave = tid >> 5;
  const int wm = wave & 1, wn = wave >> 1;
  const int row0 = blockIdx.x * 32;

  v8f acc[4] = {};

  // staging assignments
  const int ar = tid >> 3, ak = (tid & 7) * 4;  // A: b64 (4 bf16)/thread, 32x32 slab
  const int br = tid >> 2, bk = (tid & 3) * 8;  // B: 4x b128 (8 bf16)/thread, 256x32 slab
  const unsigned smb = lds_addr(smem);
  const unsigned a_st = smb + (unsigned)(ar * PD + ak) * 2;
  const unsigned b_st = smb + ABYT + (unsigned)(br * PD + bk) * 2;
  const __bf16* ap = A + (size_t)(row0 + ar) * K + ak;
  const __bf16* bp = W + (size_t)br * K + bk;

  auto stage = [&](int buf, int k0) {
    unsigned o = (unsigned)(buf * BUF);
    async_b64(a_st + o, ap + k0);
#pragma unroll
    for (int j = 0; j < 4; ++j)
      async_b128(b_st + o + (unsigned)(j * 64 * PD) * 2, bp + (size_t)j * 64 * K + k0);
  };

  stage(0, 0);
  wait_async0();
  __syncthreads();

  int cur = 0;
  const int arow = wm * 16 + (lane & 15);
  const int akb = (lane >> 4) * 8;
  const int bkb = (lane >> 4) * 8;

  for (int k0 = 0; k0 < K; k0 += 32) {
    if (k0 + 32 < K) stage(cur ^ 1, k0 + 32);  // prefetch next slab into other buffer

    const __bf16* As = (const __bf16*)(smem + cur * BUF);
    const __bf16* Bs = (const __bf16*)(smem + cur * BUF + ABYT);

    // A fragment: lane = M row; halves {0..7}=K kb.., {8..15}=K 16+kb..
    FragBF fa, fb[4];
    fa.q[0] = *(const u32x4*)&As[arow * PD + akb];
    fa.q[1] = *(const u32x4*)&As[arow * PD + 16 + akb];
#pragma unroll
    for (int j = 0; j < 4; ++j) {  // B fragment: lane = N col; halves {0..7}=K kb.., {8..15}=K 16+kb..
      const int bcol = wn * 64 + j * 16 + (lane & 15);
      fb[j].q[0] = *(const u32x4*)&Bs[bcol * PD + bkb];
      fb[j].q[1] = *(const u32x4*)&Bs[bcol * PD + 16 + bkb];
    }
#pragma unroll
    for (int j = 0; j < 4; ++j)
      acc[j] = __builtin_amdgcn_wmma_f32_16x16x32_bf16(
          false, fa.v, false, fb[j].v, (short)0, acc[j], false, false);

    wait_async0();      // my async fills of the other buffer landed
    __syncthreads();    // everyone's fills visible; everyone done reading cur
    cur ^= 1;
  }

  // spill Y (+bias) to LDS for row-wise LayerNorm (aliases staging buffers; synced above)
#pragma unroll
  for (int j = 0; j < 4; ++j) {
    int col = wn * 64 + j * 16 + (lane & 15);
    float bc = bias[col];
#pragma unroll
    for (int v = 0; v < 8; ++v) {
      int row = wm * 16 + v + ((lane >> 4) << 3);
      Ys[row * 264 + col] = acc[j][v] + bc;
    }
  }
  __syncthreads();

  // LayerNorm + L2; each wave owns 4 rows, lane covers 8 cols
  for (int rr = 0; rr < 4; ++rr) {
    int row = wave * 4 + rr;
    float y[8], s = 0.f, ss = 0.f;
#pragma unroll
    for (int i = 0; i < 8; ++i) {
      int c = lane + i * 32;
      y[i] = Ys[row * 264 + c];
      s += y[i]; ss += y[i] * y[i];
    }
#pragma unroll
    for (int off = 16; off > 0; off >>= 1) {
      s += __shfl_xor(s, off, 32);
      ss += __shfl_xor(ss, off, 32);
    }
    float mu = s * (1.f / 256.f);
    float var = ss * (1.f / 256.f) - mu * mu;
    float rstd = rsqrtf(var + 1e-5f);
    float z[8], zs = 0.f;
#pragma unroll
    for (int i = 0; i < 8; ++i) {
      int c = lane + i * 32;
      z[i] = (y[i] - mu) * rstd * gamma[c] + beta[c];
      zs += z[i] * z[i];
    }
#pragma unroll
    for (int off = 16; off > 0; off >>= 1) zs += __shfl_xor(zs, off, 32);
    float inv = 1.f / (sqrtf(zs) + 1e-6f);
    size_t base = ((size_t)(row0 + row) * 4 + mod) * Dn;
#pragma unroll
    for (int i = 0; i < 8; ++i) {
      int c = lane + i * 32;
      Zbf[base + c] = (__bf16)z[i];
      Znb[base + c] = (__bf16)(z[i] * inv);
    }
  }
}

// ---------------- NT GEMM: out[a,c] = scale * sum_d A[a,d]*Bm[c,d] ----------------
// A:[8192,256] bf16, Bm:[512,256] bf16, out fp32 [8192,512]. Block tile 64x128, async double-buffer.
__global__ __launch_bounds__(256) void gemm_nt_kernel(
    const __bf16* __restrict__ A, const __bf16* __restrict__ Bm,
    float* __restrict__ out, float scale) {
  static constexpr int ABYT = 64 * PD * 2;          // 6144
  static constexpr int BUF  = ABYT + 128 * PD * 2;  // 18432
  __shared__ __align__(16) unsigned char smem[2 * BUF];  // 36864

  const int tid = threadIdx.x, lane = tid & 31, wave = tid >> 5;
  const int wm = wave & 3, wn = wave >> 2;
  const int row0 = blockIdx.x * 64;
  const int col0 = blockIdx.y * 128;

  v8f acc[4] = {};
  const int ar = tid >> 2, ak = (tid & 3) * 8;  // b128/thread
  const unsigned smb = lds_addr(smem);
  const unsigned a_st = smb + (unsigned)(ar * PD + ak) * 2;
  const unsigned b_st = smb + ABYT + (unsigned)(ar * PD + ak) * 2;
  const __bf16* ap = A + (size_t)(row0 + ar) * Dn + ak;
  const __bf16* bp = Bm + (size_t)(col0 + ar) * Dn + ak;

  auto stage = [&](int buf, int k0) {
    unsigned o = (unsigned)(buf * BUF);
    async_b128(a_st + o, ap + k0);
    async_b128(b_st + o, bp + k0);
    async_b128(b_st + o + (unsigned)(64 * PD) * 2, bp + (size_t)64 * Dn + k0);
  };

  stage(0, 0);
  wait_async0();
  __syncthreads();

  int cur = 0;
  const int arow = wm * 16 + (lane & 15);
  const int akb = (lane >> 4) * 8;
  const int bkb = (lane >> 4) * 8;

  for (int k0 = 0; k0 < Dn; k0 += 32) {
    if (k0 + 32 < Dn) stage(cur ^ 1, k0 + 32);

    const __bf16* As = (const __bf16*)(smem + cur * BUF);
    const __bf16* Bs = (const __bf16*)(smem + cur * BUF + ABYT);

    FragBF fa, fb[4];
    fa.q[0] = *(const u32x4*)&As[arow * PD + akb];
    fa.q[1] = *(const u32x4*)&As[arow * PD + 16 + akb];
#pragma unroll
    for (int j = 0; j < 4; ++j) {
      const int bcol = wn * 64 + j * 16 + (lane & 15);
      fb[j].q[0] = *(const u32x4*)&Bs[bcol * PD + bkb];
      fb[j].q[1] = *(const u32x4*)&Bs[bcol * PD + 16 + bkb];
    }
#pragma unroll
    for (int j = 0; j < 4; ++j)
      acc[j] = __builtin_amdgcn_wmma_f32_16x16x32_bf16(
          false, fa.v, false, fb[j].v, (short)0, acc[j], false, false);

    wait_async0();
    __syncthreads();
    cur ^= 1;
  }

#pragma unroll
  for (int j = 0; j < 4; ++j) {
    int col = col0 + wn * 64 + j * 16 + (lane & 15);
#pragma unroll
    for (int v = 0; v < 8; ++v) {
      int row = row0 + wm * 16 + v + ((lane >> 4) << 3);
      out[(size_t)row * Cn + col] = acc[j][v] * scale;
    }
  }
}

// ---------------- softmax over M, contrib, logits ----------------
__global__ __launch_bounds__(256) void final_kernel(
    const float* __restrict__ sim, const float* __restrict__ OjZi,
    const float* __restrict__ cls_bias, float* __restrict__ r_out,
    float* __restrict__ contrib, float* __restrict__ logits) {
  int idx = blockIdx.x * 256 + threadIdx.x;  // over B*C
  int c = idx & (Cn - 1);
  int b = idx >> 9;
  float s[4], mx = -3.4e38f;
#pragma unroll
  for (int m = 0; m < 4; ++m) {
    s[m] = sim[((size_t)b * 4 + m) * Cn + c];
    mx = fmaxf(mx, s[m]);
  }
  float e[4], den = 0.f;
#pragma unroll
  for (int m = 0; m < 4; ++m) { e[m] = __expf(s[m] - mx); den += e[m]; }
  float inv = 1.f / den, acc = 0.f;
#pragma unroll
  for (int m = 0; m < 4; ++m) {
    size_t o = ((size_t)b * 4 + m) * Cn + c;
    float rm = e[m] * inv;
    r_out[o] = rm;
    float ct = OjZi[o] * rm;
    contrib[o] = ct;
    acc += ct;
  }
  logits[(size_t)b * Cn + c] = acc + cls_bias[c];
}

// ---------------- host launcher ----------------
extern "C" void kernel_launch(void* const* d_in, const int* in_sizes, int n_in,
                              void* d_out, int out_size, void* d_ws, size_t ws_size,
                              hipStream_t stream) {
  (void)in_sizes; (void)n_in; (void)out_size; (void)ws_size;
  const int Kdim[4] = {1024, 1024, 512, 256};
  const float *f[4], *Wm[4], *bia[4], *gam[4], *bet[4];
  for (int m = 0; m < 4; ++m) {
    f[m]   = (const float*)d_in[m * 5 + 0];
    Wm[m]  = (const float*)d_in[m * 5 + 1];
    bia[m] = (const float*)d_in[m * 5 + 2];
    gam[m] = (const float*)d_in[m * 5 + 3];
    bet[m] = (const float*)d_in[m * 5 + 4];
  }
  const float* U   = (const float*)d_in[20];
  const float* O   = (const float*)d_in[21];
  const float* cls = (const float*)d_in[22];

  char* ws = (char*)d_ws;
  size_t off = 0;
  auto take = [&](size_t bytes) -> char* {
    char* p = ws + off;
    off += (bytes + 255) & ~(size_t)255;
    return p;
  };
  __bf16* fbf[4]; for (int m = 0; m < 4; ++m) fbf[m] = (__bf16*)take((size_t)Bn * Kdim[m] * 2);
  __bf16* Wbf[4]; for (int m = 0; m < 4; ++m) Wbf[m] = (__bf16*)take((size_t)Dn * Kdim[m] * 2);
  __bf16* Unb  = (__bf16*)take((size_t)Cn * Dn * 2);
  __bf16* Obf  = (__bf16*)take((size_t)Cn * Dn * 2);
  __bf16* Zbf  = (__bf16*)take((size_t)Bn * 4 * Dn * 2);
  __bf16* Znb  = (__bf16*)take((size_t)Bn * 4 * Dn * 2);
  float*  OjZi = (float*)take((size_t)Bn * 4 * Cn * 4);

  float* out     = (float*)d_out;
  float* logits  = out;                       // [B, C]
  float* r_out   = logits + (size_t)Bn * Cn;  // [B, 4, C]
  float* sim     = r_out + (size_t)Bn * 4 * Cn;
  float* contrib = sim + (size_t)Bn * 4 * Cn;

  for (int m = 0; m < 4; ++m)
    cvt_f32_bf16<<<(Bn * Kdim[m]) / 1024, 256, 0, stream>>>(f[m], fbf[m]);
  for (int m = 0; m < 4; ++m)
    cvt_f32_bf16<<<(Dn * Kdim[m]) / 1024, 256, 0, stream>>>(Wm[m], Wbf[m]);
  cvt_f32_bf16<<<(Cn * Dn) / 1024, 256, 0, stream>>>(O, Obf);
  l2rows_kernel<<<Cn / 8, 256, 0, stream>>>(U, Unb);

  for (int m = 0; m < 4; ++m)
    proj_ln_kernel<<<Bn / 32, 256, 0, stream>>>(fbf[m], Kdim[m], Wbf[m], bia[m],
                                                gam[m], bet[m], Zbf, Znb, m);

  gemm_nt_kernel<<<dim3((Bn * 4) / 64, Cn / 128), 256, 0, stream>>>(
      Znb, Unb, sim, 1.0f / 0.07f);
  gemm_nt_kernel<<<dim3((Bn * 4) / 64, Cn / 128), 256, 0, stream>>>(
      Zbf, Obf, OjZi, 1.0f);

  final_kernel<<<(Bn * Cn) / 256, 256, 0, stream>>>(sim, OjZi, cls, r_out,
                                                    contrib, logits);
}
